// CurrentPatchEncoder_31001073943118
// MI455X (gfx1250) — compile-verified, matrix operand order checked by reading, not run on hardware
//
#include <hip/hip_runtime.h>

typedef __attribute__((ext_vector_type(16))) __bf16 v16bf;
typedef __attribute__((ext_vector_type(8)))  __bf16 v8bf;
typedef __attribute__((ext_vector_type(8)))  float  v8f;

#define D_MODEL 512
#define SEQ 16
#define NMAT (512 * 512)
#define EPSV 1.1920929e-07f
#define G 4            // patches per block
#define SCRB 53248     // per-patch scratch bytes (q16K + k16K + v16K + logits4K)

// ---- WMMA fragment helpers (CDNA5 16-bit A/B layouts, ISA 7.12.2) ----
__device__ inline v16bf frag2(const __bf16* p0, const __bf16* p1) {
  union { v16bf v; v8bf h[2]; } u;
  u.h[0] = *(const v8bf*)p0;   // K = 8*hi .. 8*hi+7
  u.h[1] = *(const v8bf*)p1;   // K = 16+8*hi .. 16+8*hi+7
  return u.v;
}

// Single 16x16 tile over K in [k0,k1): A row = lane&15 (M), B row = lane&15 (N).
__device__ inline v8f wmma_range(const __bf16* __restrict__ A, int lda,
                                 const __bf16* __restrict__ B, int ldb,
                                 int k0, int k1, int lane, v8f acc) {
  const int idx = lane & 15;
  const int ho  = 8 * (lane >> 4);
  const __bf16* arow = A + idx * lda;
  const __bf16* brow = B + idx * ldb;
  for (int kb = k0; kb < k1; kb += 32) {
    v16bf a = frag2(arow + kb + ho, arow + kb + 16 + ho);
    v16bf b = frag2(brow + kb + ho, brow + kb + 16 + ho);
    acc = __builtin_amdgcn_wmma_f32_16x16x32_bf16(false, a, false, b, (short)0, acc,
                                                  false, false);
  }
  return acc;
}

// C/D layout: lanes 0-15: M = r, N = lane; lanes 16-31: M = r+8, N = lane-16.
__device__ inline void store_tile_f32(float* out, int ldo, int n0, int lane, v8f c) {
  const int n = n0 + (lane & 15);
  const int mb = (lane >> 4) * 8;
#pragma unroll
  for (int r = 0; r < 8; ++r) out[(mb + r) * ldo + n] = c[r];
}
__device__ inline void store_tile_bf(__bf16* out, int ldo, int n0, int lane, v8f c) {
  const int n = n0 + (lane & 15);
  const int mb = (lane >> 4) * 8;
#pragma unroll
  for (int r = 0; r < 8; ++r) out[(mb + r) * ldo + n] = (__bf16)c[r];
}

// RMS-normalize G*16 rows of f32 at scr[p]+off; refresh bf16 activation mirror.
// 4 threads per row, 128 elements each.
__device__ inline void normG(unsigned char* scr, int off, __bf16* xbf,
                             float* red, float* rs, int tid) {
  const int rr = tid >> 2;              // 0..63 : (p, row)
  const int p = rr >> 4, r = rr & 15;
  const int c0 = (tid & 3) * 128;
  float* fb = (float*)(scr + (size_t)p * SCRB + off) + r * D_MODEL + c0;
  float ss = 0.f;
#pragma unroll 16
  for (int i = 0; i < 128; ++i) { float v = fb[i]; ss += v * v; }
  red[rr * 4 + (tid & 3)] = ss;
  __syncthreads();
  if (tid < 64) {
    float s = red[tid * 4] + red[tid * 4 + 1] + red[tid * 4 + 2] + red[tid * 4 + 3];
    rs[tid] = rsqrtf(s * (1.0f / D_MODEL) + EPSV);
  }
  __syncthreads();
  const float sc = rs[rr];
  __bf16* xb = xbf + p * (SEQ * D_MODEL) + r * D_MODEL + c0;
#pragma unroll 16
  for (int i = 0; i < 128; ++i) { float v = fb[i] * sc; fb[i] = v; xb[i] = (__bf16)v; }
  __syncthreads();
}

// ---- weight prep: f32 -> bf16 into workspace (L2-resident for main kernel) ----
__global__ __launch_bounds__(256) void prep_weights_kernel(
    const float* __restrict__ g, const float* __restrict__ m,
    const float* __restrict__ q, const float* __restrict__ k,
    const float* __restrict__ v, const float* __restrict__ o,
    unsigned short* __restrict__ wsu) {
  __bf16* ws = reinterpret_cast<__bf16*>(wsu);
  const int i = blockIdx.x * blockDim.x + threadIdx.x;
  if (i < NMAT) {
    ws[0 * NMAT + i] = (__bf16)g[i];
    ws[1 * NMAT + i] = (__bf16)m[i];
    ws[2 * NMAT + i] = (__bf16)q[i];
    ws[3 * NMAT + i] = (__bf16)k[i];
    ws[4 * NMAT + i] = (__bf16)v[i];
    ws[5 * NMAT + i] = (__bf16)o[i];
  }
}

// ---- main fused patch-encoder: one block (8 wave32) per G patches ----
// Each weight B-fragment read from L2 feeds G WMMAs (register reuse) -> 4x less
// L2 weight traffic than one-patch-per-block.
__global__ __launch_bounds__(256) void patch_encoder_kernel(
    const float* __restrict__ inp,   // (BP, 16, 512)
    const float* __restrict__ lpos,  // (16, 512)
    const float* __restrict__ convw, // (512, 1, 5)
    const float* __restrict__ relb,  // (4, 31)
    const float* __restrict__ wpool, // (4, 512)
    const float* __restrict__ wout,  // (512, 512) f32
    const unsigned short* __restrict__ wbits, // bf16 weights [6][512][512]
    float* __restrict__ out,         // (BP, 512)
    int npatch) {
  __shared__ __align__(16) __bf16 xbf[G * SEQ * D_MODEL];   // 64 KB activations (bf16)
  __shared__ __align__(16) unsigned char Sraw[G * SCRB];    // 208 KB phase scratch
  __shared__ float red[64 * 4];
  __shared__ float rowscale[64];
  __shared__ float plbuf[G * 16 * 4];
  __shared__ float awbuf[G * 4 * 16];
  __shared__ float pooled[G * D_MODEL];
  __shared__ float red256[256];

  const int tid  = threadIdx.x;
  const int lane = tid & 31;
  const int wave = tid >> 5;
  const int idx  = lane & 15;
  const int ho   = 8 * (lane >> 4);
  const int row  = tid >> 4, c0 = (tid & 15) * 32;   // per-patch 32-elem partition

  const __bf16* Wall = reinterpret_cast<const __bf16*>(wbits);
  const __bf16* Wgb = Wall + 0 * NMAT;
  const __bf16* Wmb = Wall + 1 * NMAT;
  const __bf16* Wqb = Wall + 2 * NMAT;
  const __bf16* Wkb = Wall + 3 * NMAT;
  const __bf16* Wvb = Wall + 4 * NMAT;
  const __bf16* Wob = Wall + 5 * NMAT;

  // ---- stage 0: x = patch + local_pos -> scr f32 ; rmsnorm -> xbf ----
#pragma unroll
  for (int p = 0; p < G; ++p) {
    int gp = blockIdx.x * G + p;
    if (gp >= npatch) gp = npatch - 1;
    const float* ip = inp + (size_t)gp * (SEQ * D_MODEL) + row * D_MODEL + c0;
    const float* lp = lpos + row * D_MODEL + c0;
    float* xp = (float*)(Sraw + (size_t)p * SCRB) + row * D_MODEL + c0;
#pragma unroll
    for (int i = 0; i < 32; ++i) xp[i] = ip[i] + lp[i];
  }
  __syncthreads();
  normG(Sraw, 0, xbf, red, rowscale, tid);

  // ---- stage 1: GLU  y = silu(x Wg^T) * (x Wm^T) -> scr f32 ----
  for (int nt = wave; nt < 32; nt += 8) {
    v8f ag[G] = {}; v8f am[G] = {};
    const __bf16* bg = Wgb + (size_t)(nt * 16 + idx) * D_MODEL;
    const __bf16* bm = Wmb + (size_t)(nt * 16 + idx) * D_MODEL;
    for (int kb = 0; kb < D_MODEL; kb += 32) {
      v16bf fbg = frag2(bg + kb + ho, bg + kb + 16 + ho);
      v16bf fbm = frag2(bm + kb + ho, bm + kb + 16 + ho);
#pragma unroll
      for (int p = 0; p < G; ++p) {
        const __bf16* ar = xbf + p * (SEQ * D_MODEL) + idx * D_MODEL;
        v16bf a = frag2(ar + kb + ho, ar + kb + 16 + ho);
        ag[p] = __builtin_amdgcn_wmma_f32_16x16x32_bf16(false, a, false, fbg, (short)0,
                                                        ag[p], false, false);
        am[p] = __builtin_amdgcn_wmma_f32_16x16x32_bf16(false, a, false, fbm, (short)0,
                                                        am[p], false, false);
      }
    }
    const int n = nt * 16 + idx, mb = (lane >> 4) * 8;
#pragma unroll
    for (int p = 0; p < G; ++p) {
      float* yF = (float*)(Sraw + (size_t)p * SCRB);
#pragma unroll
      for (int r = 0; r < 8; ++r) {
        float g = ag[p][r];
        yF[(mb + r) * D_MODEL + n] = (g / (1.f + __expf(-g))) * am[p][r];
      }
    }
  }
  __syncthreads();

  // ---- stage 2: depthwise conv1d(k=5, pad=2) residual (in place) ; rmsnorm ----
  for (int p = 0; p < G; ++p) {
    float* yF = (float*)(Sraw + (size_t)p * SCRB);
    float creg[32];
#pragma unroll 4
    for (int i = 0; i < 32; ++i) {
      const int d = c0 + i;
      float acc = yF[row * D_MODEL + d];
#pragma unroll
      for (int j = 0; j < 5; ++j) {
        const int s = row + j - 2;
        if (s >= 0 && s < SEQ) acc += yF[s * D_MODEL + d] * convw[d * 5 + j];
      }
      creg[i] = acc;
    }
    __syncthreads();
#pragma unroll 4
    for (int i = 0; i < 32; ++i) yF[row * D_MODEL + c0 + i] = creg[i];
    __syncthreads();
  }
  normG(Sraw, 0, xbf, red, rowscale, tid);

  // ---- stage 3: QKV GEMMs -> scr: q(+0) k(+16K) v(+32K), bf16 row-major ----
  for (int job = wave; job < 96; job += 8) {
    const int which = job >> 5, nt = job & 31;
    const __bf16* W = (which == 0) ? Wqb : (which == 1) ? Wkb : Wvb;
    v8f acc[G] = {};
    const __bf16* br = W + (size_t)(nt * 16 + idx) * D_MODEL;
    for (int kb = 0; kb < D_MODEL; kb += 32) {
      v16bf b = frag2(br + kb + ho, br + kb + 16 + ho);
#pragma unroll
      for (int p = 0; p < G; ++p) {
        const __bf16* ar = xbf + p * (SEQ * D_MODEL) + idx * D_MODEL;
        v16bf a = frag2(ar + kb + ho, ar + kb + 16 + ho);
        acc[p] = __builtin_amdgcn_wmma_f32_16x16x32_bf16(false, a, false, b, (short)0,
                                                         acc[p], false, false);
      }
    }
#pragma unroll
    for (int p = 0; p < G; ++p) {
      __bf16* dst = (__bf16*)(Sraw + (size_t)p * SCRB + which * 16384);
      store_tile_bf(dst, D_MODEL, nt * 16, lane, acc[p]);
    }
  }
  __syncthreads();

  // ---- stage 4: logits = q k^T / sqrt(Dh) + rel_bias (16 jobs: p x h) ----
  for (int job = wave; job < 16; job += 8) {
    const int p = job >> 2, h = job & 3;
    const __bf16* qB = (const __bf16*)(Sraw + (size_t)p * SCRB);
    const __bf16* kB = (const __bf16*)(Sraw + (size_t)p * SCRB + 16384);
    v8f acc = {};
    acc = wmma_range(qB, D_MODEL, kB, D_MODEL, h * 128, h * 128 + 128, lane, acc);
    const float isq = 0.08838834764831845f;  // 128^-0.5
    float* lg = (float*)(Sraw + (size_t)p * SCRB + 49152);
    const int n = idx, mb = (lane >> 4) * 8;
#pragma unroll
    for (int r = 0; r < 8; ++r) {
      const int s = mb + r;
      lg[h * 256 + s * 16 + n] = acc[r] * isq + relb[h * 31 + (s - n + 15)];
    }
  }
  __syncthreads();

  // ---- stage 5: softmax in place on logits (one row per thread) ----
  {
    const int p = tid >> 6, h = (tid >> 4) & 3, s = tid & 15;
    float* L = (float*)(Sraw + (size_t)p * SCRB + 49152) + h * 256 + s * 16;
    float mx = L[0];
#pragma unroll
    for (int j = 1; j < 16; ++j) mx = fmaxf(mx, L[j]);
    float e[16]; float sum = 0.f;
#pragma unroll
    for (int j = 0; j < 16; ++j) { e[j] = __expf(L[j] - mx); sum += e[j]; }
    const float inv = 1.f / sum;
#pragma unroll
    for (int j = 0; j < 16; ++j) L[j] = e[j] * inv;
  }
  __syncthreads();

  // ---- stage 6: sa = softmax @ v (K=16, VALU; ~0.5% of FLOPs) -> bf16 at +0 ----
  for (int p = 0; p < G; ++p) {
    const __bf16* vB = (const __bf16*)(Sraw + (size_t)p * SCRB + 32768);
    const float*  lg = (const float*)(Sraw + (size_t)p * SCRB + 49152);
    __bf16* sab = (__bf16*)(Sraw + (size_t)p * SCRB);   // overwrites dead q
#pragma unroll
    for (int j = 0; j < 2; ++j) {
      const int d = tid * 2 + j;
      const int h = d >> 7;
      float vf[16];
#pragma unroll
      for (int t = 0; t < 16; ++t) vf[t] = (float)vB[t * D_MODEL + d];
      const float* lh = lg + h * 256;
#pragma unroll
      for (int s = 0; s < 16; ++s) {
        float acc = 0.f;
#pragma unroll
        for (int t = 0; t < 16; ++t) acc += lh[s * 16 + t] * vf[t];
        sab[s * D_MODEL + d] = (__bf16)acc;
      }
    }
  }
  __syncthreads();

  // ---- stage 7: o = sa Wo^T -> f32 at +16K ; x = rmsnorm(x + o) ----
  for (int nt = wave; nt < 32; nt += 8) {
    v8f acc[G] = {};
    const __bf16* br = Wob + (size_t)(nt * 16 + idx) * D_MODEL;
    for (int kb = 0; kb < D_MODEL; kb += 32) {
      v16bf b = frag2(br + kb + ho, br + kb + 16 + ho);
#pragma unroll
      for (int p = 0; p < G; ++p) {
        const __bf16* ar = (const __bf16*)(Sraw + (size_t)p * SCRB) + idx * D_MODEL;
        v16bf a = frag2(ar + kb + ho, ar + kb + 16 + ho);
        acc[p] = __builtin_amdgcn_wmma_f32_16x16x32_bf16(false, a, false, b, (short)0,
                                                         acc[p], false, false);
      }
    }
#pragma unroll
    for (int p = 0; p < G; ++p)
      store_tile_f32((float*)(Sraw + (size_t)p * SCRB + 16384), D_MODEL, nt * 16, lane,
                     acc[p]);
  }
  __syncthreads();
#pragma unroll
  for (int p = 0; p < G; ++p) {   // residual: x3 = x + o  (in place at +16K)
    float* of = (float*)(Sraw + (size_t)p * SCRB + 16384) + row * D_MODEL + c0;
    const __bf16* xb = xbf + p * (SEQ * D_MODEL) + row * D_MODEL + c0;
#pragma unroll
    for (int i = 0; i < 32; ++i) of[i] += (float)xb[i];
  }
  __syncthreads();
  normG(Sraw, 16384, xbf, red, rowscale, tid);   // x3 normalized f32 stays at +16K

  // ---- stage 8: attention pooling over S ----
  {
    const int p = tid >> 6, s = (tid >> 2) & 15, hp = tid & 3;
    const float* xr = (const float*)(Sraw + (size_t)p * SCRB + 16384) + s * D_MODEL;
    const float* wp = wpool + hp * D_MODEL;
    float acc = 0.f;
    for (int d = 0; d < D_MODEL; ++d) acc += xr[d] * wp[d];
    plbuf[p * 64 + s * 4 + hp] = acc;
  }
  __syncthreads();
  if (tid < 16) {
    const int p = tid >> 2, hp = tid & 3;
    const float* pb = plbuf + p * 64;
    float mx = pb[hp];
#pragma unroll
    for (int s = 1; s < 16; ++s) mx = fmaxf(mx, pb[s * 4 + hp]);
    float e[16]; float sum = 0.f;
#pragma unroll
    for (int s = 0; s < 16; ++s) { e[s] = __expf(pb[s * 4 + hp] - mx); sum += e[s]; }
    const float inv = 1.f / sum;
#pragma unroll
    for (int s = 0; s < 16; ++s) awbuf[p * 64 + hp * 16 + s] = e[s] * inv;
  }
  __syncthreads();
#pragma unroll
  for (int p = 0; p < G; ++p) {
    const float* x3 = (const float*)(Sraw + (size_t)p * SCRB + 16384);
#pragma unroll
    for (int j = 0; j < 2; ++j) {
      const int d = tid * 2 + j;
      const float* aw = awbuf + p * 64 + (d >> 7) * 16;
      float acc = 0.f;
#pragma unroll
      for (int s = 0; s < SEQ; ++s) acc += aw[s] * x3[s * D_MODEL + d];
      pooled[p * D_MODEL + d] = acc;
    }
  }
  __syncthreads();

  // ---- stage 9: out = rmsnorm(pooled @ W_out^T) ----
  for (int p = 0; p < G; ++p) {
    const int n0 = tid * 2;
    const float* w0 = wout + (size_t)n0 * D_MODEL;
    const float* w1 = w0 + D_MODEL;
    const float* pv = pooled + p * D_MODEL;
    float a0 = 0.f, a1 = 0.f;
    for (int k = 0; k < D_MODEL; ++k) {
      const float x = pv[k];
      a0 += x * w0[k];
      a1 += x * w1[k];
    }
    red256[tid] = a0 * a0 + a1 * a1;
    __syncthreads();
    for (int off = 128; off > 0; off >>= 1) {
      if (tid < off) red256[tid] += red256[tid + off];
      __syncthreads();
    }
    const float rstd = rsqrtf(red256[0] * (1.0f / D_MODEL) + EPSV);
    const int gp = blockIdx.x * G + p;
    if (gp < npatch) {
      float* op = out + (size_t)gp * D_MODEL;
      op[n0]     = a0 * rstd;
      op[n0 + 1] = a1 * rstd;
    }
    __syncthreads();
  }
}

extern "C" void kernel_launch(void* const* d_in, const int* in_sizes, int n_in,
                              void* d_out, int out_size, void* d_ws, size_t ws_size,
                              hipStream_t stream) {
  const float* inp   = (const float*)d_in[0];
  const float* lpos  = (const float*)d_in[1];
  const float* Wg    = (const float*)d_in[2];
  const float* Wm    = (const float*)d_in[3];
  const float* convw = (const float*)d_in[4];
  const float* Wq    = (const float*)d_in[5];
  const float* Wk    = (const float*)d_in[6];
  const float* Wv    = (const float*)d_in[7];
  const float* Wo    = (const float*)d_in[8];
  const float* relb  = (const float*)d_in[9];
  const float* wpool = (const float*)d_in[10];
  const float* wout  = (const float*)d_in[11];
  unsigned short* wsbf = (unsigned short*)d_ws;

  prep_weights_kernel<<<(NMAT + 255) / 256, 256, 0, stream>>>(Wg, Wm, Wq, Wk, Wv, Wo, wsbf);

  const int patches = in_sizes[0] / (SEQ * D_MODEL);  // 8192
  const int blocks = (patches + G - 1) / G;           // 2048
  patch_encoder_kernel<<<blocks, 256, 0, stream>>>(inp, lpos, convw, relb, wpool, wout,
                                                   wsbf, (float*)d_out, patches);
}
